// S4Layer_9423158248016
// MI455X (gfx1250) — compile-verified, matrix operand order checked by reading, not run on hardware
//
#include <hip/hip_runtime.h>
#include <math.h>
#include <stdint.h>

#define DEVI __device__ __forceinline__

// Problem constants (match reference)
constexpr int  Bsz = 8, Hc = 256, Lc = 4096, Nst = 32, Gg = 32;
constexpr int  CPG = Hc / Gg;              // 8 channels per group
constexpr int  NC  = 8,  Tc = Lc / NC;     // 8 chunks of 512
constexpr long long BL = (long long)Bsz * Lc;  // 32768 positions

typedef __attribute__((ext_vector_type(16))) __bf16 bf16x16;
typedef __attribute__((ext_vector_type(8)))  float  floatx8;

// ---------------------------------------------------------------------------
// CDNA5 async Global->LDS copy (ASYNCcnt-tracked) with builtin/asm fallback
// ---------------------------------------------------------------------------
#define GAS __attribute__((address_space(1)))
#define LAS __attribute__((address_space(3)))
typedef int v4i __attribute__((vector_size(4 * sizeof(int))));

DEVI void async_ld_b128(const void* g, void* l) {
#if __has_builtin(__builtin_amdgcn_global_load_async_to_lds_b128)
  __builtin_amdgcn_global_load_async_to_lds_b128(
      (GAS v4i*)(g), (LAS v4i*)(l), 0, 0);
#else
  unsigned loff = (unsigned)(uintptr_t)l;
  asm volatile("global_load_async_to_lds_b128 %0, %1, off"
               :: "v"(loff), "v"(g) : "memory");
#endif
}

DEVI void wait_async0() {
#if __has_builtin(__builtin_amdgcn_s_wait_asynccnt)
  __builtin_amdgcn_s_wait_asynccnt(0);
#else
  asm volatile("s_wait_asynccnt 0x0" ::: "memory");
#endif
  asm volatile("" ::: "memory");
}
DEVI void wait_async1() {
#if __has_builtin(__builtin_amdgcn_s_wait_asynccnt)
  __builtin_amdgcn_s_wait_asynccnt(1);
#else
  asm volatile("s_wait_asynccnt 0x1" ::: "memory");
#endif
  asm volatile("" ::: "memory");
}

// Stage 128 consecutive floats (one wave, 4 floats / lane) into LDS async.
DEVI void async_tile128(const float* gsrc, float* ldst) {
  int lane = threadIdx.x & 31;
  async_ld_b128(gsrc + lane * 4, ldst + lane * 4);
}

// ---------------------------------------------------------------------------
// WMMA helpers
// ---------------------------------------------------------------------------
DEVI floatx8 wmma_bf16(bf16x16 a, bf16x16 b, floatx8 c) {
  return __builtin_amdgcn_wmma_f32_16x16x32_bf16(false, a, false, b, (short)0, c,
                                                 false, false);
}

// K index held by fragment element e for the documented 16-bit A/B layout.
DEVI int wmma_k_of(int e, int half) {
  int r = e >> 1, s = e & 1;
  return (r < 4 ? (r << 1) : 16 + ((r - 4) << 1)) + s + (half << 3);
}

DEVI bf16x16 load_a_frag(const __bf16* As, int rowBase, int lda) {
  int lane = threadIdx.x & 31;
  int m = lane & 15, half = lane >> 4;
  bf16x16 f;
#pragma unroll
  for (int e = 0; e < 16; ++e)
    f[e] = As[(rowBase + m) * lda + wmma_k_of(e, half)];
  return f;
}

DEVI bf16x16 load_bT_frag(const __bf16* BsT, int colBase, int ld) {
  int lane = threadIdx.x & 31;
  int n = lane & 15, half = lane >> 4;
  bf16x16 f;
#pragma unroll
  for (int e = 0; e < 16; ++e)
    f[e] = BsT[(colBase + n) * ld + wmma_k_of(e, half)];
  return f;
}

// ---------------------------------------------------------------------------
// K1: GroupNorm statistics -> per-(b,h) affine scale/shift
// ---------------------------------------------------------------------------
__global__ void k_gn_stats(const float* __restrict__ x,
                           const float* __restrict__ gn_w,
                           const float* __restrict__ gn_b,
                           float* __restrict__ scale, float* __restrict__ shift) {
  int bg = blockIdx.x;
  int b = bg / Gg, g = bg % Gg;
  const float* xp = x + ((size_t)b * Hc + (size_t)g * CPG) * Lc;
  const int n = CPG * Lc;
  int tid = threadIdx.x;
  float s = 0.f, ss = 0.f;
  for (int i = tid; i < n; i += 256) { float v = xp[i]; s += v; ss += v * v; }
  __shared__ float r0[256], r1[256];
  r0[tid] = s; r1[tid] = ss; __syncthreads();
  for (int st = 128; st > 0; st >>= 1) {
    if (tid < st) { r0[tid] += r0[tid + st]; r1[tid] += r1[tid + st]; }
    __syncthreads();
  }
  __shared__ float mu_s, rstd_s;
  if (tid == 0) {
    float mu = r0[0] / (float)n;
    float var = r1[0] / (float)n - mu * mu;
    mu_s = mu; rstd_s = rsqrtf(var + 1e-5f);
  }
  __syncthreads();
  if (tid < CPG) {
    int h = g * CPG + tid;
    float sc = rstd_s * gn_w[h];
    scale[b * Hc + h] = sc;
    shift[b * Hc + h] = gn_b[h] - mu_s * sc;
  }
}

// ---------------------------------------------------------------------------
// K2: discretized SSM parameters per (h,n)
// ---------------------------------------------------------------------------
__global__ void k_params(const float* __restrict__ log_dt,
                         const float* __restrict__ A_log_re,
                         const float* __restrict__ A_im,
                         const float* __restrict__ C_re,
                         const float* __restrict__ C_im,
                         float4* __restrict__ ab,   // (a_re, a_im, 2Cd_re, -2Cd_im)
                         float2* __restrict__ aT) {
  int idx = blockIdx.x * blockDim.x + threadIdx.x;
  if (idx >= Hc * Nst) return;
  int h = idx >> 5;
  float dt  = expf(log_dt[h]);
  float Are = -expf(A_log_re[idx]);
  float Aim = A_im[idx];
  float er  = expf(Are * dt);
  float are = er * cosf(Aim * dt);
  float aim = er * sinf(Aim * dt);
  float dre = are - 1.f, dim = aim;
  float inv = 1.f / (Are * Are + Aim * Aim);
  float qre = (dre * Are + dim * Aim) * inv;
  float qim = (dim * Are - dre * Aim) * inv;
  float Cr = C_re[idx], Ci = C_im[idx];
  float cdre = Cr * qre - Ci * qim;
  float cdim = Cr * qim + Ci * qre;
  ab[idx] = make_float4(are, aim, 2.f * cdre, -2.f * cdim);
  float tr = are, ti = aim;
#pragma unroll
  for (int i = 0; i < 9; ++i) { float nr = tr*tr - ti*ti, ni = 2.f*tr*ti; tr = nr; ti = ni; }
  aT[idx] = make_float2(tr, ti);
}

// ---------------------------------------------------------------------------
// K3: per-(b,h,chunk) chunk-final state; x streamed via async Global->LDS
// ---------------------------------------------------------------------------
__global__ void __launch_bounds__(256)
k_chunk_state(const float* __restrict__ x,
              const float* __restrict__ scale,
              const float* __restrict__ shift,
              const float4* __restrict__ ab,
              float2* __restrict__ cstate) {
  __shared__ __align__(16) float xbuf[8][2][128];   // per-wave double buffer
  int wave = (blockIdx.x * blockDim.x + threadIdx.x) >> 5;
  int lane = threadIdx.x & 31;
  int wv = threadIdx.x >> 5;
  if (wave >= Bsz * Hc * NC) return;
  int c = wave & (NC - 1);
  int bh = wave / NC;
  int h = bh & (Hc - 1);
  float4 p = ab[h * Nst + lane];
  float sc = scale[bh], sh = shift[bh];
  const float* xp = x + (size_t)bh * Lc + (size_t)c * Tc;
  float* buf = &xbuf[wv][0][0];
  float sr = 0.f, si = 0.f;

  async_tile128(xp, buf);                       // prime tile 0
  constexpr int NS = Tc / 128;                  // 4 sub-tiles
  for (int s = 0; s < NS; ++s) {
    if (s + 1 < NS) { async_tile128(xp + (s + 1) * 128, buf + ((s + 1) & 1) * 128); wait_async1(); }
    else            { wait_async0(); }
    const float* cur = buf + (s & 1) * 128;
    for (int t = 0; t < 128; ++t) {
      float u = cur[t] * sc + sh;               // LDS broadcast read
      float nr = p.x * sr - p.y * si + u;
      float ni = p.x * si + p.y * sr;
      sr = nr; si = ni;
    }
  }
  cstate[(size_t)wave * Nst + lane] = make_float2(sr, si);
}

// ---------------------------------------------------------------------------
// K4: exclusive scan over chunks -> chunk initial states
// ---------------------------------------------------------------------------
__global__ void k_chunk_scan(const float2* __restrict__ aT,
                             const float2* __restrict__ cstate,
                             float2* __restrict__ cinit) {
  int wave = (blockIdx.x * blockDim.x + threadIdx.x) >> 5;
  int lane = threadIdx.x & 31;
  if (wave >= Bsz * Hc) return;
  int h = wave & (Hc - 1);
  float2 t = aT[h * Nst + lane];
  float cr = 0.f, ci = 0.f;
  for (int c = 0; c < NC; ++c) {
    size_t o = ((size_t)wave * NC + c) * Nst + lane;
    cinit[o] = make_float2(cr, ci);
    float2 s = cstate[o];
    float nr = t.x * cr - t.y * ci + s.x;
    float ni = t.x * ci + t.y * cr + s.y;
    cr = nr; ci = ni;
  }
}

// ---------------------------------------------------------------------------
// K5: full recurrence; fused D-skip + exact GELU; async-streamed x
// ---------------------------------------------------------------------------
__global__ void __launch_bounds__(256)
k_ssm_out(const float* __restrict__ x,
          const float* __restrict__ scale,
          const float* __restrict__ shift,
          const float4* __restrict__ ab,
          const float2* __restrict__ cinit,
          const float* __restrict__ Dv,
          __bf16* __restrict__ u) {
  __shared__ __align__(16) float xbuf[8][2][128];
  int wave = (blockIdx.x * blockDim.x + threadIdx.x) >> 5;
  int lane = threadIdx.x & 31;
  int wv = threadIdx.x >> 5;
  if (wave >= Bsz * Hc * NC) return;
  int c = wave & (NC - 1);
  int bh = wave / NC;
  int h = bh & (Hc - 1);
  int b = bh / Hc;
  float4 p = ab[h * Nst + lane];
  float sc = scale[bh], sh = shift[bh];
  float Dh = Dv[h];
  float2 s0 = cinit[(size_t)wave * Nst + lane];
  float sr = s0.x, si = s0.y;
  const float* xp = x + (size_t)bh * Lc + (size_t)c * Tc;
  __bf16* up = u + (size_t)h * BL + (size_t)b * Lc + (size_t)c * Tc;
  float* buf = &xbuf[wv][0][0];

  async_tile128(xp, buf);
  constexpr int NS = Tc / 128;
  for (int s = 0; s < NS; ++s) {
    if (s + 1 < NS) { async_tile128(xp + (s + 1) * 128, buf + ((s + 1) & 1) * 128); wait_async1(); }
    else            { wait_async0(); }
    const float* cur = buf + (s & 1) * 128;
    for (int t0 = 0; t0 < 128; t0 += 32) {
      float mine = 0.f;
#pragma unroll 4
      for (int k = 0; k < 32; ++k) {
        float uin = cur[t0 + k] * sc + sh;
        float nr = p.x * sr - p.y * si + uin;
        float ni = p.x * si + p.y * sr;
        sr = nr; si = ni;
        float part = p.z * sr + p.w * si;        // 2*Re(Cd * s)
#pragma unroll
        for (int off = 16; off > 0; off >>= 1) part += __shfl_xor(part, off, 32);
        float y = part + Dh * uin;
        float gel = 0.5f * y * (1.f + erff(y * 0.70710678f));
        if (k == lane) mine = gel;
      }
      up[s * 128 + t0 + lane] = (__bf16)mine;
    }
  }
}

// ---------------------------------------------------------------------------
// K6: GLU projection, bf16 WMMA; both halves of W_out share the B-tile.
// ---------------------------------------------------------------------------
__global__ void __launch_bounds__(256)
k_glu_gemm(const float* __restrict__ W, const float* __restrict__ bias,
           const __bf16* __restrict__ u, __bf16* __restrict__ g) {
  __shared__ __bf16 Asm[2][64][34];
  __shared__ __bf16 BsT[128][34];
  int n0 = blockIdx.x * 128;
  int m0 = blockIdx.y * 64;
  int tid = threadIdx.x;
  int wv = tid >> 5;
  int wm = wv & 3, wn = wv >> 2;

  floatx8 accT[4] = {{}, {}, {}, {}};
  floatx8 accB[4] = {{}, {}, {}, {}};

  for (int kk = 0; kk < Hc; kk += 32) {
    for (int idx = tid; idx < 64 * 32; idx += 256) {
      int row = idx >> 5, k = idx & 31;
      Asm[0][row][k] = (__bf16)W[(size_t)(m0 + row) * Hc + kk + k];
      Asm[1][row][k] = (__bf16)W[(size_t)(m0 + 256 + row) * Hc + kk + k];
    }
    for (int idx = tid; idx < 32 * 128; idx += 256) {
      int k = idx >> 7, n = idx & 127;
      BsT[n][k] = u[(size_t)(kk + k) * BL + n0 + n];
    }
    if (kk + 32 < Hc)   // prefetch next B K-tile -> global_prefetch_b8
      __builtin_prefetch(&u[(size_t)(kk + 32 + (tid & 31)) * BL + n0], 0, 1);
    __syncthreads();

    bf16x16 at = load_a_frag(&Asm[0][0][0], wm * 16, 34);
    bf16x16 ab = load_a_frag(&Asm[1][0][0], wm * 16, 34);
#pragma unroll
    for (int j = 0; j < 4; ++j) {
      bf16x16 bf = load_bT_frag(&BsT[0][0], wn * 64 + j * 16, 34);
      accT[j] = wmma_bf16(at, bf, accT[j]);
      accB[j] = wmma_bf16(ab, bf, accB[j]);
    }
    __syncthreads();
  }

  int lane = tid & 31;
  int nl = lane & 15, half = lane >> 4;
#pragma unroll
  for (int j = 0; j < 4; ++j) {
#pragma unroll
    for (int r = 0; r < 8; ++r) {
      int row = (half << 3) + r;
      int gm = m0 + wm * 16 + row;
      size_t col = (size_t)n0 + wn * 64 + j * 16 + nl;
      float zt = accT[j][r] + bias[gm];
      float zb = accB[j][r] + bias[gm + 256];
      float gv = zt * (1.f / (1.f + expf(-zb)));
      g[(size_t)gm * BL + col] = (__bf16)gv;
    }
  }
}

// ---------------------------------------------------------------------------
// K7: conv1d(k=3,pad=1) as 3 shifted-tap WMMA GEMMs; fused bias + residual.
// ---------------------------------------------------------------------------
__global__ void __launch_bounds__(256)
k_conv_gemm(const float* __restrict__ cw,
            const float* __restrict__ cb,
            const __bf16* __restrict__ g,
            const float* __restrict__ x,
            float* __restrict__ out) {
  __shared__ __bf16 Asm[3][64][34];
  __shared__ __bf16 BsT[130][34];
  int n0 = blockIdx.x * 128;
  int m0 = blockIdx.y * 64;
  int tid = threadIdx.x;
  int wv = tid >> 5;
  int wm = wv & 3, wn = wv >> 2;
  int bstart = (n0 / Lc) * Lc;

  floatx8 acc[4] = {{}, {}, {}, {}};

  for (int kk = 0; kk < Hc; kk += 32) {
    for (int idx = tid; idx < 3 * 64 * 32; idx += 256) {
      int d = idx >> 11;
      int rem = idx & 2047;
      int row = rem >> 5, k = rem & 31;
      Asm[d][row][k] = (__bf16)cw[(size_t)(m0 + row) * (Hc * 3) + (kk + k) * 3 + d];
    }
    for (int idx = tid; idx < 32 * 130; idx += 256) {
      int k = idx / 130;
      int j = idx - k * 130;
      long long nn = (long long)n0 - 1 + j;
      __bf16 v = (__bf16)0.0f;
      if (nn >= bstart && nn < bstart + Lc)
        v = g[(size_t)(kk + k) * BL + nn];
      BsT[j][k] = v;
    }
    __syncthreads();

#pragma unroll
    for (int d = 0; d < 3; ++d) {
      bf16x16 af = load_a_frag(&Asm[d][0][0], wm * 16, 34);
#pragma unroll
      for (int j = 0; j < 4; ++j) {
        bf16x16 bf = load_bT_frag(&BsT[0][0], wn * 64 + j * 16 + d, 34);
        acc[j] = wmma_bf16(af, bf, acc[j]);
      }
    }
    __syncthreads();
  }

  int lane = tid & 31;
  int nl = lane & 15, half = lane >> 4;
#pragma unroll
  for (int j = 0; j < 4; ++j) {
#pragma unroll
    for (int r = 0; r < 8; ++r) {
      int row = (half << 3) + r;
      int o = m0 + wm * 16 + row;
      long long ng = (long long)n0 + wn * 64 + j * 16 + nl;
      int b = (int)(ng >> 12);
      int l = (int)(ng & (Lc - 1));
      size_t oi = ((size_t)b * Hc + o) * Lc + l;
      out[oi] = x[oi] + cb[o] + acc[j][r];
    }
  }
}

// ---------------------------------------------------------------------------
extern "C" void kernel_launch(void* const* d_in, const int* in_sizes, int n_in,
                              void* d_out, int out_size, void* d_ws, size_t ws_size,
                              hipStream_t stream) {
  const float* x        = (const float*)d_in[0];
  const float* gn_w     = (const float*)d_in[1];
  const float* gn_b     = (const float*)d_in[2];
  const float* log_dt   = (const float*)d_in[3];
  const float* A_log_re = (const float*)d_in[4];
  const float* A_im     = (const float*)d_in[5];
  const float* C_re     = (const float*)d_in[6];
  const float* C_im     = (const float*)d_in[7];
  const float* Dv       = (const float*)d_in[8];
  const float* W_out    = (const float*)d_in[9];
  const float* b_out    = (const float*)d_in[10];
  const float* conv_w   = (const float*)d_in[11];
  const float* conv_b   = (const float*)d_in[12];
  float* out = (float*)d_out;

  char* ws = (char*)d_ws;
  size_t off = 0;
  auto alloc = [&](size_t bytes) { size_t o = off; off = (off + bytes + 255) & ~(size_t)255; return o; };
  float*  w_scale  = (float*) (ws + alloc(sizeof(float)  * Bsz * Hc));
  float*  w_shift  = (float*) (ws + alloc(sizeof(float)  * Bsz * Hc));
  float4* w_ab     = (float4*)(ws + alloc(sizeof(float4) * Hc * Nst));
  float2* w_aT     = (float2*)(ws + alloc(sizeof(float2) * Hc * Nst));
  float2* w_cstate = (float2*)(ws + alloc(sizeof(float2) * (size_t)Bsz * Hc * NC * Nst));
  float2* w_cinit  = (float2*)(ws + alloc(sizeof(float2) * (size_t)Bsz * Hc * NC * Nst));
  __bf16* w_u      = (__bf16*)(ws + alloc(sizeof(__bf16) * (size_t)Hc * BL));
  __bf16* w_g      = (__bf16*)(ws + alloc(sizeof(__bf16) * (size_t)Hc * BL));

  k_gn_stats<<<Bsz * Gg, 256, 0, stream>>>(x, gn_w, gn_b, w_scale, w_shift);
  k_params<<<(Hc * Nst + 255) / 256, 256, 0, stream>>>(log_dt, A_log_re, A_im,
                                                       C_re, C_im, w_ab, w_aT);
  k_chunk_state<<<(Bsz * Hc * NC) / 8, 256, 0, stream>>>(x, w_scale, w_shift,
                                                         w_ab, w_cstate);
  k_chunk_scan<<<(Bsz * Hc) / 8, 256, 0, stream>>>(w_aT, w_cstate, w_cinit);
  k_ssm_out<<<(Bsz * Hc * NC) / 8, 256, 0, stream>>>(x, w_scale, w_shift, w_ab,
                                                     w_cinit, Dv, w_u);
  dim3 gemm_grid((unsigned)(BL / 128), Hc / 64);
  k_glu_gemm<<<gemm_grid, 256, 0, stream>>>(W_out, b_out, w_u, w_g);
  k_conv_gemm<<<gemm_grid, 256, 0, stream>>>(conv_w, conv_b, w_g, x, out);
}
